// InterpretableMultiHeadAttention_90323162235390
// MI455X (gfx1250) — matrix operand order repaired and checked by reading, not compiled, not run on hardware
//
#include <hip/hip_runtime.h>
#include <stdint.h>

#define HIDDEN 1024
#define HEADS  16
#define DATTN  64
#define BATCH  2
#define QLEN   2048
#define KLEN   2048

typedef __bf16 v16bf __attribute__((ext_vector_type(16)));
typedef float  v8f   __attribute__((ext_vector_type(8)));

union FragU { uint4 u[2]; v16bf v; };

__device__ __forceinline__ uint32_t pack_bf16x2(float a, float b) {
    uint32_t ua = __float_as_uint(a);
    uint32_t ub = __float_as_uint(b);
    ua = (ua + 0x7FFFu + ((ua >> 16) & 1u)) >> 16;   // RNE
    ub = (ub + 0x7FFFu + ((ub >> 16) & 1u)) >> 16;
    return (ua & 0xFFFFu) | (ub << 16);
}

__device__ __forceinline__ uint16_t f32_to_bf16(float f) {
    uint32_t u = __float_as_uint(f);
    u = (u + 0x7FFFu + ((u >> 16) & 1u)) >> 16;
    return (uint16_t)u;
}

__device__ __forceinline__ uint4 cvt_chunk(const float* __restrict__ p) {
    float4 f0 = *(const float4*)p;
    float4 f1 = *(const float4*)(p + 4);
    uint4 w;
    w.x = pack_bf16x2(f0.x, f0.y);
    w.y = pack_bf16x2(f0.z, f0.w);
    w.z = pack_bf16x2(f1.x, f1.y);
    w.w = pack_bf16x2(f1.z, f1.w);
    return w;
}

// CDNA5 async global->LDS 128-bit copy (ASYNCcnt tracked, no VGPR data).
__device__ __forceinline__ void async_copy_b128(uint32_t lds_byte_off, const void* gptr) {
    uint64_t ga = (uint64_t)(uintptr_t)gptr;
    asm volatile("global_load_async_to_lds_b128 %0, %1, off"
                 :: "v"(lds_byte_off), "v"(ga) : "memory");
}
__device__ __forceinline__ void wait_async0() {
    asm volatile("s_wait_asynccnt 0x0" ::: "memory");
}

// MODE: 0 = f32 store, 1 = f32 store + bias, 2 = f32 transposed store,
//       3 = bf16 store (D is uint16_t*).
// D = scale * (A @ Bt^T), A[M][K], Bt[N][K] f32 row-major along K; converted
// to bf16 in LDS, f32 accumulate via v_wmma_f32_16x16x32_bf16.
// Block tile 128x64, 8 wave32, each wave owns 2x2 16x16 tiles (4 WMMA/K-step).
template <int MODE>
__global__ __launch_bounds__(256) void gemm_f32in_wmma(
    const float* __restrict__ A,  long lda, long aStrideO, long aStrideI,
    const float* __restrict__ Bt, long ldb, long bStrideO, long bStrideI,
    void* __restrict__ Dv,        long ldd, long dStrideO, long dStrideI,
    int K, int zInner, float scale, const float* __restrict__ bias)
{
    __shared__ uint32_t sA[128 * 16];  // 128 rows x 32 bf16
    __shared__ uint32_t sB[64 * 16];   // 64 N-rows x 32 bf16

    const int tid  = threadIdx.x;
    const int lane = tid & 31;
    const int wave = tid >> 5;

    const int zo = blockIdx.z / zInner;
    const int zi = blockIdx.z % zInner;
    A  += (long)zo * aStrideO + (long)zi * aStrideI;
    Bt += (long)zo * bStrideO + (long)zi * bStrideI;

    const int  ldRow = tid >> 2;
    const int  ldKc  = (tid & 3) * 8;
    const long gm0 = (long)blockIdx.x * 128 + ldRow;
    const long gm1 = gm0 + 64;
    const long gn  = (long)blockIdx.y * 64 + ldRow;

    const int mt0 = (wave & 3) * 2;
    const int nt0 = (wave >> 2) * 2;

    v8f c00 = {}, c01 = {}, c10 = {}, c11 = {};

    const uint32_t* pA0 = sA + ((mt0 * 16) + (lane & 15)) * 16 + ((lane < 16) ? 0 : 4);
    const uint32_t* pA1 = pA0 + 16 * 16;
    const uint32_t* pB0 = sB + ((nt0 * 16) + (lane & 15)) * 16 + ((lane < 16) ? 0 : 4);
    const uint32_t* pB1 = pB0 + 16 * 16;

    for (int k0 = 0; k0 < K; k0 += 32) {
        __syncthreads();
        *(uint4*)(sA + ldRow * 16 + (tid & 3) * 4)        = cvt_chunk(A + gm0 * lda + (k0 + ldKc));
        *(uint4*)(sA + (ldRow + 64) * 16 + (tid & 3) * 4) = cvt_chunk(A + gm1 * lda + (k0 + ldKc));
        *(uint4*)(sB + ldRow * 16 + (tid & 3) * 4)        = cvt_chunk(Bt + gn  * ldb + (k0 + ldKc));
        __syncthreads();

        FragU a0, a1, b0, b1;
        a0.u[0] = *(const uint4*)pA0;  a0.u[1] = *(const uint4*)(pA0 + 8);
        a1.u[0] = *(const uint4*)pA1;  a1.u[1] = *(const uint4*)(pA1 + 8);
        b0.u[0] = *(const uint4*)pB0;  b0.u[1] = *(const uint4*)(pB0 + 8);
        b1.u[0] = *(const uint4*)pB1;  b1.u[1] = *(const uint4*)(pB1 + 8);

        c00 = __builtin_amdgcn_wmma_f32_16x16x32_bf16(false, a0.v, false, b0.v, (short)0, c00, false, false);
        c01 = __builtin_amdgcn_wmma_f32_16x16x32_bf16(false, a0.v, false, b1.v, (short)0, c01, false, false);
        c10 = __builtin_amdgcn_wmma_f32_16x16x32_bf16(false, a1.v, false, b0.v, (short)0, c10, false, false);
        c11 = __builtin_amdgcn_wmma_f32_16x16x32_bf16(false, a1.v, false, b1.v, (short)0, c11, false, false);
    }

    #pragma unroll
    for (int i = 0; i < 2; ++i) {
        #pragma unroll
        for (int j = 0; j < 2; ++j) {
            v8f acc = (i == 0) ? ((j == 0) ? c00 : c01) : ((j == 0) ? c10 : c11);
            const long m0c = (long)blockIdx.x * 128 + (mt0 + i) * 16 + ((lane < 16) ? 0 : 8);
            const long col = (long)blockIdx.y * 64 + (nt0 + j) * 16 + (lane & 15);
            if (MODE == 2) {
                float* D = (float*)Dv + (long)zo * dStrideO + (long)zi * dStrideI;
                float4 lo, hi;
                lo.x = acc[0] * scale; lo.y = acc[1] * scale;
                lo.z = acc[2] * scale; lo.w = acc[3] * scale;
                hi.x = acc[4] * scale; hi.y = acc[5] * scale;
                hi.z = acc[6] * scale; hi.w = acc[7] * scale;
                float* dp = D + col * ldd + m0c;
                *(float4*)dp       = lo;
                *(float4*)(dp + 4) = hi;
            } else if (MODE == 3) {
                uint16_t* D = (uint16_t*)Dv + (long)zo * dStrideO + (long)zi * dStrideI;
                #pragma unroll
                for (int r = 0; r < 8; ++r)
                    D[(m0c + r) * ldd + col] = f32_to_bf16(acc[r] * scale);
            } else {
                float* D = (float*)Dv + (long)zo * dStrideO + (long)zi * dStrideI;
                const float bv = (MODE == 1) ? bias[col] : 0.0f;
                #pragma unroll
                for (int r = 0; r < 8; ++r)
                    D[(m0c + r) * ldd + col] = acc[r] * scale + bv;
            }
        }
    }
}

// Scores GEMM: A[M][K] and Bt[N][K] already bf16 -> stage tiles into LDS with
// async 128-bit global->LDS copies (no VALU in the loader), 4 WMMA per K-step.
__global__ __launch_bounds__(256) void gemm_bf16in_wmma_async(
    const uint16_t* __restrict__ A,  long lda, long aStrideO, long aStrideI,
    const uint16_t* __restrict__ Bt, long ldb, long bStrideO, long bStrideI,
    float* __restrict__ D,           long ldd, long dStrideO, long dStrideI,
    int K, int zInner, float scale)
{
    __shared__ uint32_t sA[128 * 16];
    __shared__ uint32_t sB[64 * 16];

    const int tid  = threadIdx.x;
    const int lane = tid & 31;
    const int wave = tid >> 5;

    const int zo = blockIdx.z / zInner;
    const int zi = blockIdx.z % zInner;
    A  += (long)zo * aStrideO + (long)zi * aStrideI;
    Bt += (long)zo * bStrideO + (long)zi * bStrideI;
    D  += (long)zo * dStrideO + (long)zi * dStrideI;

    const int  ldRow = tid >> 2;            // 0..63
    const int  ldKc  = (tid & 3) * 8;       // bf16 elements: 0,8,16,24
    const long gm0 = (long)blockIdx.x * 128 + ldRow;
    const long gm1 = gm0 + 64;
    const long gn  = (long)blockIdx.y * 64 + ldRow;

    // LDS byte offsets (flat-address low 32 bits carry the LDS offset)
    const uint32_t sAoff = (uint32_t)(uintptr_t)sA;
    const uint32_t sBoff = (uint32_t)(uintptr_t)sB;
    const uint32_t dstA0 = sAoff + (uint32_t)(ldRow * 64 + (tid & 3) * 16);
    const uint32_t dstA1 = dstA0 + 64u * 64u;
    const uint32_t dstB  = sBoff + (uint32_t)(ldRow * 64 + (tid & 3) * 16);

    const int mt0 = (wave & 3) * 2;
    const int nt0 = (wave >> 2) * 2;

    v8f c00 = {}, c01 = {}, c10 = {}, c11 = {};

    const uint32_t* pA0 = sA + ((mt0 * 16) + (lane & 15)) * 16 + ((lane < 16) ? 0 : 4);
    const uint32_t* pA1 = pA0 + 16 * 16;
    const uint32_t* pB0 = sB + ((nt0 * 16) + (lane & 15)) * 16 + ((lane < 16) ? 0 : 4);
    const uint32_t* pB1 = pB0 + 16 * 16;

    for (int k0 = 0; k0 < K; k0 += 32) {
        __syncthreads();
        async_copy_b128(dstA0, A + gm0 * lda + (k0 + ldKc));
        async_copy_b128(dstA1, A + gm1 * lda + (k0 + ldKc));
        async_copy_b128(dstB,  Bt + gn * ldb + (k0 + ldKc));
        wait_async0();
        __syncthreads();

        FragU a0, a1, b0, b1;
        a0.u[0] = *(const uint4*)pA0;  a0.u[1] = *(const uint4*)(pA0 + 8);
        a1.u[0] = *(const uint4*)pA1;  a1.u[1] = *(const uint4*)(pA1 + 8);
        b0.u[0] = *(const uint4*)pB0;  b0.u[1] = *(const uint4*)(pB0 + 8);
        b1.u[0] = *(const uint4*)pB1;  b1.u[1] = *(const uint4*)(pB1 + 8);

        c00 = __builtin_amdgcn_wmma_f32_16x16x32_bf16(false, a0.v, false, b0.v, (short)0, c00, false, false);
        c01 = __builtin_amdgcn_wmma_f32_16x16x32_bf16(false, a0.v, false, b1.v, (short)0, c01, false, false);
        c10 = __builtin_amdgcn_wmma_f32_16x16x32_bf16(false, a1.v, false, b0.v, (short)0, c10, false, false);
        c11 = __builtin_amdgcn_wmma_f32_16x16x32_bf16(false, a1.v, false, b1.v, (short)0, c11, false, false);
    }

    #pragma unroll
    for (int i = 0; i < 2; ++i) {
        #pragma unroll
        for (int j = 0; j < 2; ++j) {
            v8f acc = (i == 0) ? ((j == 0) ? c00 : c01) : ((j == 0) ? c10 : c11);
            const long m0c = (long)blockIdx.x * 128 + (mt0 + i) * 16 + ((lane < 16) ? 0 : 8);
            const long col = (long)blockIdx.y * 64 + (nt0 + j) * 16 + (lane & 15);
            #pragma unroll
            for (int r = 0; r < 8; ++r)
                D[(m0c + r) * ldd + col] = acc[r] * scale;
        }
    }
}

// Per (b,q): softmax each head's score row in place (-> attn_weights output)
// and write the head-averaged probabilities to the avg workspace.
__global__ __launch_bounds__(256) void softmax_avg_kernel(
    float* __restrict__ attn, float* __restrict__ avg)
{
    __shared__ float red[256];
    const int tid = threadIdx.x;
    const int bq  = blockIdx.x;
    const int b   = bq / QLEN;
    const int q   = bq % QLEN;

    float acc[8];
    #pragma unroll
    for (int i = 0; i < 8; ++i) acc[i] = 0.0f;

    for (int h = 0; h < HEADS; ++h) {
        float* row = attn + (((long)(b * HEADS + h) * QLEN + q) * KLEN);
        float x[8];
        float m = -3.4e38f;
        #pragma unroll
        for (int i = 0; i < 8; ++i) { x[i] = row[tid + i * 256]; m = fmaxf(m, x[i]); }
        red[tid] = m; __syncthreads();
        for (int s = 128; s > 0; s >>= 1) {
            if (tid < s) red[tid] = fmaxf(red[tid], red[tid + s]);
            __syncthreads();
        }
        m = red[0]; __syncthreads();

        float sum = 0.0f;
        #pragma unroll
        for (int i = 0; i < 8; ++i) { x[i] = __expf(x[i] - m); sum += x[i]; }
        red[tid] = sum; __syncthreads();
        for (int s = 128; s > 0; s >>= 1) {
            if (tid < s) red[tid] += red[tid + s];
            __syncthreads();
        }
        const float inv = 1.0f / red[0]; __syncthreads();

        #pragma unroll
        for (int i = 0; i < 8; ++i) {
            float p = x[i] * inv;
            row[tid + i * 256] = p;
            acc[i] += p;
        }
    }

    float* arow = avg + (long)bq * KLEN;
    const float invH = 1.0f / (float)HEADS;
    #pragma unroll
    for (int i = 0; i < 8; ++i) arow[tid + i * 256] = acc[i] * invH;
}

extern "C" void kernel_launch(void* const* d_in, const int* in_sizes, int n_in,
                              void* d_out, int out_size, void* d_ws, size_t ws_size,
                              hipStream_t stream)
{
    const float* query = (const float*)d_in[0];
    const float* key   = (const float*)d_in[1];
    const float* value = (const float*)d_in[2];
    const float* w_q   = (const float*)d_in[3];
    const float* w_k   = (const float*)d_in[4];
    const float* w_v   = (const float*)d_in[5];
    const float* w_out = (const float*)d_in[6];
    const float* b_out = (const float*)d_in[7];

    float* out  = (float*)d_out;                          // [B,QLEN,HIDDEN]
    float* attn = out + (long)BATCH * QLEN * HIDDEN;      // [B,H,QLEN,KLEN]

    char* wsb = (char*)d_ws;
    uint16_t* Qb = (uint16_t*)wsb;                                   // bf16 [B*QLEN,HIDDEN]
    uint16_t* Kb = (uint16_t*)(wsb + (size_t)BATCH*QLEN*HIDDEN*2);   // bf16 [B*KLEN,HIDDEN]
    float* Vt  = (float*)(wsb + (size_t)BATCH*QLEN*HIDDEN*4);        // [DATTN, B*KLEN] (V^T)
    float* ctx = Vt + (long)DATTN * BATCH * KLEN;                    // [B*QLEN, DATTN]
    float* avg = ctx + (long)BATCH * QLEN * DATTN;                   // [B*QLEN, KLEN]

    const dim3 blk(256);
    const int  MR = BATCH * QLEN;   // 4096

    // Q (bf16 out) = query @ w_q^T
    gemm_f32in_wmma<3><<<dim3(MR/128, HIDDEN/64, 1), blk, 0, stream>>>(
        query, HIDDEN, 0, 0,  w_q, HIDDEN, 0, 0,  Qb, HIDDEN, 0, 0,
        HIDDEN, 1, 1.0f, nullptr);
    // K (bf16 out) = key @ w_k^T
    gemm_f32in_wmma<3><<<dim3(MR/128, HIDDEN/64, 1), blk, 0, stream>>>(
        key, HIDDEN, 0, 0,    w_k, HIDDEN, 0, 0,  Kb, HIDDEN, 0, 0,
        HIDDEN, 1, 1.0f, nullptr);
    // V^T[d][b*KLEN+k] = (value @ w_v^T)[b,k,d]
    gemm_f32in_wmma<2><<<dim3(MR/128, DATTN/64, 1), blk, 0, stream>>>(
        value, HIDDEN, 0, 0,  w_v, HIDDEN, 0, 0,  Vt, (long)BATCH*KLEN, 0, 0,
        HIDDEN, 1, 1.0f, nullptr);
    // scores[b,h] = (Qh @ Kh^T)/8, bf16 inputs staged with async b128 copies
    gemm_bf16in_wmma_async<<<dim3(QLEN/128, KLEN/64, BATCH*HEADS), blk, 0, stream>>>(
        Qb, HIDDEN, (long)QLEN*HIDDEN, DATTN,
        Kb, HIDDEN, (long)KLEN*HIDDEN, DATTN,
        attn, KLEN, (long)HEADS*QLEN*KLEN, (long)QLEN*KLEN,
        DATTN, HEADS, 0.125f);
    // softmax in-place + head average
    softmax_avg_kernel<<<dim3(MR), blk, 0, stream>>>(attn, avg);
    // context[b] = avg[b] @ V[b]
    gemm_f32in_wmma<0><<<dim3(QLEN/128, DATTN/64, BATCH), blk, 0, stream>>>(
        avg, KLEN, (long)QLEN*KLEN, 0,
        Vt, (long)BATCH*KLEN, (long)KLEN, 0,
        ctx, DATTN, (long)QLEN*DATTN, 0,
        KLEN, 1, 1.0f, nullptr);
    // output = ctx @ w_out^T + b_out
    gemm_f32in_wmma<1><<<dim3(MR/128, HIDDEN/64, 1), blk, 0, stream>>>(
        ctx, DATTN, 0, 0,  w_out, DATTN, 0, 0,  out, HIDDEN, 0, 0,
        DATTN, 1, 1.0f, b_out);
}
